// Attention_DHOGSA_21062519619987
// MI455X (gfx1250) — matrix operand order repaired and checked by reading, not compiled, not run on hardware
//
#include <hip/hip_runtime.h>
#include <hip/hip_bf16.h>

// ---------------------------------------------------------------------------
// Types for CDNA5 WMMA (wave32): v_wmma_f32_16x16x32_bf16
// ---------------------------------------------------------------------------
typedef __attribute__((ext_vector_type(16))) __bf16 bf16x16;
typedef __attribute__((ext_vector_type(8)))  float  f32x8;
typedef __attribute__((ext_vector_type(4)))  unsigned u32x4;
typedef __attribute__((ext_vector_type(8)))  int i32x8;
typedef __attribute__((ext_vector_type(4)))  int i32x4;

union FragAB { bf16x16 bf; u32x4 q[2]; };

#define DEV static __device__ __forceinline__

constexpr int kB = 2, kC = 64, kHalf = 32, kH = 256, kW = 256;
constexpr int kHW = kH * kW;                 // 65536
constexpr int kHeads = 8, kCph = 8, kF = 8;
constexpr int kHWh = kHW / kF;               // 8192
constexpr int kQKVC = 5 * kC;                // 320
constexpr int kNP = 32, kNPatch = kNP * kNP; // 1024 patches of 8x8
constexpr int kNBins = 9;
constexpr float kPI = 3.14159265358979323846f;

DEV unsigned short f2bf(float f) {
  unsigned u = __float_as_uint(f);
  u += 0x7FFFu + ((u >> 16) & 1u);           // round-to-nearest-even
  return (unsigned short)(u >> 16);
}
DEV float bf2f(unsigned short h) { return __uint_as_float(((unsigned)h) << 16); }

// A fragment: 16(M) x 32(K) bf16, rows of LDS tile [m][k], row stride lda (ushorts).
// Lane layout per ISA 7.12.2: lanes 0-15 hold K {0..7,16..23}, lanes 16-31 {8..15,24..31}.
DEV bf16x16 fragA(const unsigned short* s, int lda, int m0, int k0, int ll, int hi) {
  FragAB f;
  const unsigned short* row = s + (m0 + ll) * lda;
  f.q[0] = *reinterpret_cast<const u32x4*>(row + k0 + hi * 8);
  f.q[1] = *reinterpret_cast<const u32x4*>(row + k0 + 16 + hi * 8);
  return f.bf;
}
// B fragment: 32(K) x 16(N) bf16, staged TRANSPOSED in LDS as [n][k].
// Lanes 0-15 hold K {0..7,16..23}, lanes 16-31 {8..15,24..31} (matches A).
DEV bf16x16 fragB(const unsigned short* s, int ldb, int n0, int k0, int ll, int hi) {
  FragAB f;
  const unsigned short* row = s + (n0 + ll) * ldb;
  f.q[0] = *reinterpret_cast<const u32x4*>(row + k0 + hi * 8);
  f.q[1] = *reinterpret_cast<const u32x4*>(row + k0 + 16 + hi * 8);
  return f.bf;
}
DEV f32x8 wmma_bf16(bf16x16 a, bf16x16 b, f32x8 c) {
  return __builtin_amdgcn_wmma_f32_16x16x32_bf16(false, a, false, b, (short)0, c,
                                                 false, false);
}

// ---------------------------------------------------------------------------
// Tensor Data Mover: DMA a 2D bf16 tile (tileD0 contiguous elems per line,
// tileD1 lines, line stride strideElems) from global into LDS at ldsOff.
// D# layout per CDNA5 ISA (group0 + group1, 2D tensor,
// groups 2/3 zero). Issued per-wave; caller waits s_wait_tensorcnt.
// ---------------------------------------------------------------------------
DEV void tdm_load_2d(unsigned ldsOff, const unsigned short* gsrc, unsigned tileD0,
                     unsigned tileD1, unsigned long long strideElems) {
  unsigned long long ga = (unsigned long long)(size_t)gsrc;
  u32x4 g0;
  g0.x = 1u;                                    // count=1 (valid), user mode
  g0.y = ldsOff;                                // lds_addr (bytes)
  g0.z = (unsigned)ga;                          // global_addr[31:0]
  g0.w = (unsigned)((ga >> 32) & 0x01FFFFFFull) // global_addr[56:32]
         | (2u << 30);                          // type=2 ("image")
  i32x8 g1;
  g1[0] = 0x00010000;                           // mask=0, data_size=1 (2 bytes)
  g1[1] = (int)((tileD0 & 0xFFFFu) << 16);      // tensor_dim0[15:0]
  g1[2] = (int)(((tileD0 >> 16) & 0xFFFFu) | ((tileD1 & 0xFFFFu) << 16));
  g1[3] = (int)(((tileD1 >> 16) & 0xFFFFu) | ((tileD0 & 0xFFFFu) << 16)); // tile_dim0
  g1[4] = (int)(tileD1 & 0xFFFFu);              // tile_dim1 (tile_dim2=0)
  g1[5] = (int)(strideElems & 0xFFFFFFFFull);   // tensor_dim0_stride[31:0]
  g1[6] = (int)((strideElems >> 32) & 0xFFFFull);
  g1[7] = 0;
  i32x4 z4 = (i32x4)0;
#if __clang_major__ >= 23
  __builtin_amdgcn_tensor_load_to_lds(g0, g1, z4, z4, (i32x8)0, 0);
#else
  __builtin_amdgcn_tensor_load_to_lds(g0, g1, z4, z4, 0);
#endif
}
DEV void tdm_wait() { __builtin_amdgcn_s_wait_tensorcnt(0); }
DEV unsigned lds_off(const void* p) { return (unsigned)(size_t)p; }  // addr[31:0] = LDS offset

// ---------------------------------------------------------------------------
// Workspace layout (bytes, 256-aligned)
// ---------------------------------------------------------------------------
constexpr size_t AL(size_t x) { return (x + 255) & ~(size_t)255; }
constexpr size_t SZ_F32_HALF = (size_t)kB * kHalf * kHW * 4;       // 16 MB
constexpr size_t SZ_SIDXP    = (size_t)kB * kNPatch * 64 * 4;
constexpr size_t SZ_HOG      = (size_t)kB * kNPatch * kNBins * 4;
constexpr size_t SZ_HMS      = (size_t)kB * kHalf * kNPatch * 4;
constexpr size_t SZ_BF_FULL  = (size_t)kB * kC * kHW * 2;          // 16 MB
constexpr size_t SZ_WQKVBF   = (size_t)kQKVC * kC * 2;
constexpr size_t SZ_WPROJBF  = (size_t)kC * kC * 2;
constexpr size_t SZ_QKV      = (size_t)kB * kQKVC * kHW * 4;       // 167 MB
constexpr size_t SZ_KEYS     = (size_t)kB * kHW * 4;
constexpr size_t SZ_RNORM    = 4 * 1024 * 4;
constexpr size_t SZ_ABF      = (size_t)2 * kB * kHeads * 4096 * 2;
constexpr size_t SZ_OF32     = (size_t)kB * kC * kHW * 4;          // 33 MB

constexpr size_t OFF_MAGH   = 0;
constexpr size_t OFF_BINH   = OFF_MAGH + AL(SZ_F32_HALF);
constexpr size_t OFF_SIDXP  = OFF_BINH + AL(SZ_F32_HALF);
constexpr size_t OFF_HOG    = OFF_SIDXP + AL(SZ_SIDXP);
constexpr size_t OFF_HMS    = OFF_HOG + AL(SZ_HOG);
constexpr size_t OFF_XHP    = OFF_HMS + AL(SZ_HMS);
constexpr size_t OFF_X2BF   = OFF_XHP + AL(SZ_F32_HALF);
constexpr size_t OFF_WQKVB  = OFF_X2BF + AL(SZ_BF_FULL);
constexpr size_t OFF_WPROJB = OFF_WQKVB + AL(SZ_WQKVBF);
constexpr size_t OFF_QKV1   = OFF_WPROJB + AL(SZ_WPROJBF);
constexpr size_t OFF_QKV2   = OFF_QKV1 + AL(SZ_QKV);
constexpr size_t OFF_KEYS   = OFF_QKV2 + AL(SZ_QKV);
constexpr size_t OFF_IDX    = OFF_KEYS + AL(SZ_KEYS);
constexpr size_t OFF_GQ1    = OFF_IDX + AL(SZ_KEYS);
constexpr size_t OFF_GK1    = OFF_GQ1 + AL(SZ_BF_FULL);
constexpr size_t OFF_GQ2    = OFF_GK1 + AL(SZ_BF_FULL);
constexpr size_t OFF_GK2    = OFF_GQ2 + AL(SZ_BF_FULL);
constexpr size_t OFF_GV     = OFF_GK2 + AL(SZ_BF_FULL);
constexpr size_t OFF_RNORM  = OFF_GV + AL(SZ_BF_FULL);
constexpr size_t OFF_ABF    = OFF_RNORM + AL(SZ_RNORM);
// The qkv1 buffer retires after the depthwise conv; reuse its space:
constexpr size_t OFF_O1     = OFF_QKV1;
constexpr size_t OFF_O2     = OFF_O1 + AL(SZ_OF32);
constexpr size_t OFF_MBF    = OFF_O2 + AL(SZ_OF32);
constexpr size_t OFF_OUTT   = OFF_MBF + AL(SZ_BF_FULL);
static_assert(OFF_OUTT + SZ_OF32 <= OFF_QKV2, "alias region overflow");

// ---------------------------------------------------------------------------
// Stage 1: Sobel fields (mag, orientation bin) on first 32 channels
// ---------------------------------------------------------------------------
__global__ void k_sobel_hog(const float* __restrict__ x, float* __restrict__ mag,
                            int* __restrict__ bin) {
  int e = blockIdx.x * blockDim.x + threadIdx.x;
  if (e >= kB * kHalf * kHW) return;
  int p = e % kHW, c = (e / kHW) % kHalf, b = e / (kHW * kHalf);
  int yy = p >> 8, xx = p & 255;
  const float* src = x + (size_t)(b * kC + c) * kHW;
  auto at = [&](int y2, int x2) -> float {
    return (y2 < 0 || y2 >= kH || x2 < 0 || x2 >= kW) ? 0.f : src[y2 * kW + x2];
  };
  float a00 = at(yy - 1, xx - 1), a01 = at(yy - 1, xx), a02 = at(yy - 1, xx + 1);
  float a10 = at(yy, xx - 1), a12 = at(yy, xx + 1);
  float a20 = at(yy + 1, xx - 1), a21 = at(yy + 1, xx), a22 = at(yy + 1, xx + 1);
  float gx = (a02 - a00) + 2.f * (a12 - a10) + (a22 - a20);
  float gy = (a20 - a00) + 2.f * (a21 - a01) + (a22 - a02);
  float m = sqrtf(gx * gx + gy * gy + 1e-6f);
  float ori = atan2f(gy, gx);
  int bi = (int)((ori + kPI) / (2.f * kPI) * 9.f) % 9;
  mag[e] = m;
  bin[e] = bi;
}

// ---------------------------------------------------------------------------
// Stage 2: per-patch HOG + 64-element bitonic argsort + patch-pixel gather
// ---------------------------------------------------------------------------
__global__ void k_patch(const float* __restrict__ x, const float* __restrict__ mag,
                        const int* __restrict__ bin, float* __restrict__ xhp,
                        int* __restrict__ sidxp, float* __restrict__ hog) {
  __shared__ float s_key[64];
  __shared__ int s_idx[64];
  __shared__ float s_hog[kNBins];
  int n = blockIdx.x % kNPatch, b = blockIdx.x / kNPatch;
  int t = threadIdx.x;
  int ph = n >> 5, pw = n & 31;
  int yy = ph * 8 + (t >> 3), xx = pw * 8 + (t & 7);
  if (t < kNBins) s_hog[t] = 0.f;
  __syncthreads();
  float kv = 0.f;
  for (int c = 0; c < kHalf; ++c) {
    size_t off = (size_t)(b * kHalf + c) * kHW + yy * kW + xx;
    float m = mag[off];
    int bi = bin[off];
    kv += m * (float)(bi + 1);
    atomicAdd(&s_hog[bi], m);
  }
  s_key[t] = kv;
  s_idx[t] = t;
  __syncthreads();
  for (int k = 2; k <= 64; k <<= 1)
    for (int j = k >> 1; j > 0; j >>= 1) {
      int ixj = t ^ j;
      if (ixj > t) {
        bool up = ((t & k) == 0);
        float ka = s_key[t], kb = s_key[ixj];
        if ((ka > kb) == up) {
          s_key[t] = kb; s_key[ixj] = ka;
          int tmp = s_idx[t]; s_idx[t] = s_idx[ixj]; s_idx[ixj] = tmp;
        }
      }
      __syncthreads();
    }
  sidxp[((size_t)b * kNPatch + n) * 64 + t] = s_idx[t];
  if (t == 0) {
    float hs = 0.f, hv[kNBins];
    for (int i = 0; i < kNBins; ++i) { hv[i] = s_hog[i] * (1.f / 2048.f); hs += hv[i]; }
    float inv = 1.f / (hs + 1e-8f);
    for (int i = 0; i < kNBins; ++i)
      hog[((size_t)b * kNPatch + n) * kNBins + i] = hv[i] * inv;
  }
  int u = s_idx[t];
  int sy = ph * 8 + (u >> 3), sx = pw * 8 + (u & 7);
  for (int c = 0; c < kHalf; ++c)
    xhp[(size_t)(b * kHalf + c) * kHW + yy * kW + xx] =
        x[(size_t)(b * kC + c) * kHW + sy * kW + sx];
}

// 1x1 conv over bins: hog (b,1024,9) -> hogmap_small (b,32,32,32)
__global__ void k_binconv(const float* __restrict__ hog, const float* __restrict__ wbin,
                          float* __restrict__ hms) {
  int e = blockIdx.x * blockDim.x + threadIdx.x;
  if (e >= kB * kHalf * kNPatch) return;
  int pp = e % kNPatch, oc = (e / kNPatch) % kHalf, b = e / (kNPatch * kHalf);
  const float* h = hog + ((size_t)b * kNPatch + pp) * kNBins;
  float acc = 0.f;
  for (int i = 0; i < kNBins; ++i) acc += wbin[oc * kNBins + i] * h[i];
  hms[e] = acc;
}

// x2 = concat(xh_p + bilinear_up(hogmap), x[:,32:]) converted to bf16
__global__ void k_buildx2(const float* __restrict__ x, const float* __restrict__ xhp,
                          const float* __restrict__ hms, unsigned short* __restrict__ x2) {
  int e = blockIdx.x * blockDim.x + threadIdx.x;
  if (e >= kB * kC * kHW) return;
  int p = e % kHW, c = (e / kHW) % kC, b = e / (kHW * kC);
  float v;
  if (c < kHalf) {
    int yy = p >> 8, xx = p & 255;
    float fy = (yy + 0.5f) * 0.125f - 0.5f, fx = (xx + 0.5f) * 0.125f - 0.5f;
    int y0 = (int)floorf(fy), x0 = (int)floorf(fx);
    float wy = fy - y0, wx = fx - x0;
    int y0c = min(max(y0, 0), 31), y1c = min(max(y0 + 1, 0), 31);
    int x0c = min(max(x0, 0), 31), x1c = min(max(x0 + 1, 0), 31);
    const float* hm = hms + (size_t)(b * kHalf + c) * kNPatch;
    float hv = hm[y0c * 32 + x0c] * (1 - wy) * (1 - wx) +
               hm[y0c * 32 + x1c] * (1 - wy) * wx +
               hm[y1c * 32 + x0c] * wy * (1 - wx) + hm[y1c * 32 + x1c] * wy * wx;
    v = xhp[(size_t)(b * kHalf + c) * kHW + p] + hv;
  } else {
    v = x[(size_t)(b * kC + c) * kHW + p];
  }
  x2[e] = f2bf(v);
}

__global__ void k_cvtw(const float* __restrict__ wq, const float* __restrict__ wp,
                       unsigned short* __restrict__ wqb, unsigned short* __restrict__ wpb) {
  int i = blockIdx.x * blockDim.x + threadIdx.x;
  if (i < kQKVC * kC) wqb[i] = f2bf(wq[i]);
  if (i < kC * kC) wpb[i] = f2bf(wp[i]);
}

// ---------------------------------------------------------------------------
// Generic 1x1-conv GEMM: Y[b][m][s] = sum_k W[m][k] * X[b][k][s], K=64
// A-tile (contiguous 8KB of weights) is DMA'd by the TDM while all threads
// stage the transposed B-tile; WMMA bf16 with f32 accumulation.
// ---------------------------------------------------------------------------
__global__ void k_gemm64(const unsigned short* __restrict__ Wbf,
                         const unsigned short* __restrict__ Xbf, float* __restrict__ Y,
                         size_t xBatch, size_t yBatch) {
  __shared__ __align__(16) unsigned short As[64 * 64];
  __shared__ __align__(16) unsigned short Bs[64 * 72];
  int tid = threadIdx.x;
  int m0 = blockIdx.y * 64, s0 = blockIdx.x * 64;
  size_t bo = (size_t)blockIdx.z * xBatch;
  if (tid < 32)  // wave 0: TDM the whole 64x64 bf16 A tile (one contiguous line)
    tdm_load_2d(lds_off(As), Wbf + (size_t)m0 * kC, 4096, 1, 4096);
  for (int e = tid; e < 4096; e += 256) {
    int n = e & 63, k = e >> 6;
    Bs[n * 72 + k] = Xbf[bo + (size_t)k * kHW + s0 + n];  // transposed stage
  }
  if (tid < 32) tdm_wait();
  __syncthreads();
  int wave = tid >> 5, lane = tid & 31, ll = lane & 15, hi = lane >> 4;
  for (int tt = 0; tt < 2; ++tt) {
    int tile = wave * 2 + tt, mi = tile >> 2, ni = tile & 3;
    f32x8 acc = (f32x8)0.0f;
#pragma unroll
    for (int k0 = 0; k0 < 64; k0 += 32)
      acc = wmma_bf16(fragA(As, 64, mi * 16, k0, ll, hi),
                      fragB(Bs, 72, ni * 16, k0, ll, hi), acc);
#pragma unroll
    for (int r = 0; r < 8; ++r) {
      int row = m0 + mi * 16 + r + hi * 8;
      int col = s0 + ni * 16 + ll;
      Y[(size_t)blockIdx.z * yBatch + (size_t)row * kHW + col] = acc[r];
    }
  }
}

// depthwise 3x3 pad 1 on all 320 channels
__global__ void k_dw(const float* __restrict__ in, const float* __restrict__ wdw,
                     float* __restrict__ outp) {
  int e = blockIdx.x * blockDim.x + threadIdx.x;
  if (e >= kB * kQKVC * kHW) return;
  int p = e % kHW, ch = (e / kHW) % kQKVC, b = e / (kHW * kQKVC);
  int yy = p >> 8, xx = p & 255;
  const float* src = in + (size_t)(b * kQKVC + ch) * kHW;
  const float* w = wdw + ch * 9;
  float acc = 0.f;
#pragma unroll
  for (int ky = 0; ky < 3; ++ky)
#pragma unroll
    for (int kx = 0; kx < 3; ++kx) {
      int y2 = yy + ky - 1, x2 = xx + kx - 1;
      if (y2 >= 0 && y2 < kH && x2 >= 0 && x2 < kW)
        acc += w[ky * 3 + kx] * src[y2 * kW + x2];
    }
  outp[e] = acc;
}

// sort keys: sum_c |grad(v)| * normalized-orientation
__global__ void k_keys(const float* __restrict__ qkv2, float* __restrict__ keys,
                       int* __restrict__ idxb) {
  int e = blockIdx.x * blockDim.x + threadIdx.x;
  if (e >= kB * kHW) return;
  int p = e % kHW, b = e / kHW;
  int yy = p >> 8, xx = p & 255;
  const float* vbase = qkv2 + ((size_t)b * kQKVC + 4 * kC) * kHW;
  float s = 0.f;
  for (int c = 0; c < kC; ++c) {
    const float* src = vbase + (size_t)c * kHW;
    auto at = [&](int y2, int x2) -> float {
      return (y2 < 0 || y2 >= kH || x2 < 0 || x2 >= kW) ? 0.f : src[y2 * kW + x2];
    };
    float a00 = at(yy - 1, xx - 1), a01 = at(yy - 1, xx), a02 = at(yy - 1, xx + 1);
    float a10 = at(yy, xx - 1), a12 = at(yy, xx + 1);
    float a20 = at(yy + 1, xx - 1), a21 = at(yy + 1, xx), a22 = at(yy + 1, xx + 1);
    float gx = (a02 - a00) + 2.f * (a12 - a10) + (a22 - a20);
    float gy = (a20 - a00) + 2.f * (a21 - a01) + (a22 - a02);
    float m = sqrtf(gx * gx + gy * gy + 1e-6f);
    float o = (atan2f(gy, gx) + kPI) * (0.5f / kPI);
    s += m * o;
  }
  keys[e] = s;
  idxb[e] = p;
}

__global__ void k_bitonic(float* __restrict__ keys, int* __restrict__ idxb, int j, int k) {
  int t = blockIdx.x * blockDim.x + threadIdx.x;
  if (t >= kB * kHW) return;
  int i = t & (kHW - 1), b = t >> 16;
  int ixj = i ^ j;
  if (ixj <= i) return;
  size_t gi = (size_t)b * kHW + i, gj = (size_t)b * kHW + ixj;
  float ka = keys[gi], kb = keys[gj];
  if ((ka > kb) == ((i & k) == 0)) {
    keys[gi] = kb; keys[gj] = ka;
    int tmp = idxb[gi]; idxb[gi] = idxb[gj]; idxb[gj] = tmp;
  }
}

// gather q1,k1,q2,k2,v along hw with idx, converting to bf16
__global__ void k_gather(const float* __restrict__ qkv2, const int* __restrict__ idxb,
                         unsigned short* __restrict__ g0, unsigned short* __restrict__ g1,
                         unsigned short* __restrict__ g2, unsigned short* __restrict__ g3,
                         unsigned short* __restrict__ g4) {
  int e = blockIdx.x * blockDim.x + threadIdx.x;
  if (e >= kB * kC * kHW) return;
  int j = e % kHW, c = (e / kHW) % kC, b = e / (kHW * kC);
  int s = idxb[(size_t)b * kHW + j];
  size_t w = (size_t)(b * kC + c) * kHW + j;
  size_t base = ((size_t)b * kQKVC + c) * kHW + s;
  g0[w] = f2bf(qkv2[base + (size_t)0 * kC * kHW]);
  g1[w] = f2bf(qkv2[base + (size_t)1 * kC * kHW]);
  g2[w] = f2bf(qkv2[base + (size_t)2 * kC * kHW]);
  g3[w] = f2bf(qkv2[base + (size_t)3 * kC * kHW]);
  g4[w] = f2bf(qkv2[base + (size_t)4 * kC * kHW]);
}

// reciprocal L2 row norms for qh/kh of each branch (fold l2norm into epilogue)
__global__ void k_norms(const unsigned short* __restrict__ gq1,
                        const unsigned short* __restrict__ gk1,
                        const unsigned short* __restrict__ gq2,
                        const unsigned short* __restrict__ gk2,
                        float* __restrict__ rnorm) {
  int wid = (blockIdx.x * blockDim.x + threadIdx.x) >> 5;
  int lane = threadIdx.x & 31;
  if (wid >= 4096) return;
  int row = wid & 63, rest = wid >> 6;
  int head = rest & 7, b = (rest >> 3) & 1, which = (rest >> 4) & 3;
  const unsigned short* g = which == 0 ? gq1 : which == 1 ? gk1 : which == 2 ? gq2 : gk2;
  bool box = which < 2;
  size_t chBase = (size_t)(b * kC + head * kCph + (row >> 3)) * kHW;
  int f_ = row & 7;
  float s = 0.f;
  for (int d = lane; d < kHWh; d += 32) {
    size_t off = chBase + (box ? (size_t)f_ * kHWh + d : (size_t)d * kF + f_);
    float v = bf2f(g[off]);
    s += v * v;
  }
  for (int m = 16; m > 0; m >>= 1) s += __shfl_xor(s, m, 32);
  if (lane == 0)
    rnorm[which * 1024 + (b * 8 + head) * 64 + row] = 1.f / fmaxf(sqrtf(s), 1e-12f);
}

// ---------------------------------------------------------------------------
// Attention QK^T: 64x64 over K=8192 per (branch,b,head), + scaled softmax_1.
// Box branch rows have a UNIFORM 8192-elem stride (ch stride 65536 = 8*8192),
// so Q/K 32x64 tiles are DMA'd by the TDM (wave0=Q, wave1=K) each K-step.
// ---------------------------------------------------------------------------
__global__ void k_attn_qk(const unsigned short* __restrict__ gq1,
                          const unsigned short* __restrict__ gk1,
                          const unsigned short* __restrict__ gq2,
                          const unsigned short* __restrict__ gk2,
                          const float* __restrict__ rnorm, const float* __restrict__ temp,
                          unsigned short* __restrict__ abf) {
  __shared__ __align__(16) unsigned short Qs[64 * 32];
  __shared__ __align__(16) unsigned short Ks[64 * 32];
  __shared__ float a_s[64 * 68];
  int head = blockIdx.x & 7, b = blockIdx.x >> 3, branch = blockIdx.y;
  const unsigned short* gq = branch ? gq2 : gq1;
  const unsigned short* gk = branch ? gk2 : gk1;
  bool box = (branch == 0);
  int tid = threadIdx.x, wave = tid >> 5, lane = tid & 31, ll = lane & 15, hi = lane >> 4;
  size_t hb = (size_t)(b * kC + head * kCph) * kHW;
  f32x8 acc0 = (f32x8)0.0f, acc1 = (f32x8)0.0f;
  for (int it = 0; it < kHWh / 32; ++it) {
    int k0 = it * 32;
    if (box) {
      if (wave == 0) {         // DMA Q tile: 32 elems/line, 64 lines, stride 8192
        tdm_load_2d(lds_off(Qs), gq + hb + k0, 32, 64, (unsigned long long)kHWh);
        tdm_wait();
      } else if (wave == 1) {  // DMA K tile
        tdm_load_2d(lds_off(Ks), gk + hb + k0, 32, 64, (unsigned long long)kHWh);
        tdm_wait();
      }
    } else {
      for (int e = tid; e < 2048; e += 256) {
        int r = e >> 5, kk = e & 31;
        size_t off = hb + (size_t)(r >> 3) * kHW + (size_t)(k0 + kk) * kF + (r & 7);
        Qs[r * 32 + kk] = gq[off];
        Ks[r * 32 + kk] = gk[off];
      }
      if (it + 1 < kHWh / 32) {  // prefetch next K-tile (global_prefetch_b8)
        int r = tid >> 2, kk2 = (tid & 3) * 8;
        size_t off = hb + (size_t)(r >> 3) * kHW + (size_t)(k0 + 32 + kk2) * kF + (r & 7);
        __builtin_prefetch(gq + off, 0, 0);
        __builtin_prefetch(gk + off, 0, 0);
      }
    }
    __syncthreads();
    {
      int tile = wave * 2, mi = tile >> 2, ni = tile & 3;
      acc0 = wmma_bf16(fragA(Qs, 32, mi * 16, 0, ll, hi),
                       fragB(Ks, 32, ni * 16, 0, ll, hi), acc0);
      tile = wave * 2 + 1; mi = tile >> 2; ni = tile & 3;
      acc1 = wmma_bf16(fragA(Qs, 32, mi * 16, 0, ll, hi),
                       fragB(Ks, 32, ni * 16, 0, ll, hi), acc1);
    }
    __syncthreads();
  }
  for (int tt = 0; tt < 2; ++tt) {
    int tile = wave * 2 + tt, mi = tile >> 2, ni = tile & 3;
    f32x8 a = tt ? acc1 : acc0;
#pragma unroll
    for (int r = 0; r < 8; ++r) a_s[(mi * 16 + r + hi * 8) * 68 + ni * 16 + ll] = a[r];
  }
  __syncthreads();
  if (tid < 64) {
    int r = tid, nb = (b * 8 + head) * 64;
    float rq = rnorm[branch * 2048 + nb + r];
    const float* rkp = rnorm + (branch * 2 + 1) * 1024 + nb;
    float tp = temp[head];
    float sum = 0.f;
    for (int c2 = 0; c2 < 64; ++c2) {
      float ev = expf(a_s[r * 68 + c2] * rq * rkp[c2] * tp);
      a_s[r * 68 + c2] = ev;
      sum += ev;
    }
    float inv = 1.f / (sum + 1.f);  // softmax_1
    unsigned short* ab = abf + ((size_t)branch * 16 + b * 8 + head) * 4096 + r * 64;
    for (int c2 = 0; c2 < 64; ++c2) ab[c2] = f2bf(a_s[r * 68 + c2] * inv);
  }
}

// O = A(64x64) * V(64x8192), one 64-wide N tile per block.
// The contiguous 8KB attention matrix is DMA'd by the TDM.
__global__ void k_attn_o(const unsigned short* __restrict__ abf,
                         const unsigned short* __restrict__ gv, float* __restrict__ o1,
                         float* __restrict__ o2) {
  __shared__ __align__(16) unsigned short As[64 * 64];
  __shared__ __align__(16) unsigned short Bs[64 * 72];
  int d0 = blockIdx.x * 64;
  int head = blockIdx.y & 7, b = blockIdx.y >> 3, branch = blockIdx.z;
  bool box = (branch == 0);
  float* ob = branch ? o2 : o1;
  int tid = threadIdx.x, wave = tid >> 5, lane = tid & 31, ll = lane & 15, hi = lane >> 4;
  const unsigned short* ab = abf + ((size_t)branch * 16 + b * 8 + head) * 4096;
  size_t hb = (size_t)(b * kC + head * kCph) * kHW;
  if (tid < 32) tdm_load_2d(lds_off(As), ab, 4096, 1, 4096);
  for (int e = tid; e < 4096; e += 256) {
    int d = e & 63, j = e >> 6;
    size_t off = hb + (size_t)(j >> 3) * kHW +
                 (box ? (size_t)(j & 7) * kHWh + (d0 + d)
                      : (size_t)(d0 + d) * kF + (j & 7));
    Bs[d * 72 + j] = gv[off];  // transposed stage [d][j]
  }
  if (tid < 32) tdm_wait();
  __syncthreads();
  for (int tt = 0; tt < 2; ++tt) {
    int tile = wave * 2 + tt, mi = tile >> 2, ni = tile & 3;
    f32x8 acc = (f32x8)0.0f;
#pragma unroll
    for (int k0 = 0; k0 < 64; k0 += 32)
      acc = wmma_bf16(fragA(As, 64, mi * 16, k0, ll, hi),
                      fragB(Bs, 72, ni * 16, k0, ll, hi), acc);
#pragma unroll
    for (int r = 0; r < 8; ++r) {
      int i = mi * 16 + r + hi * 8;
      int dc = d0 + ni * 16 + ll;
      size_t off = hb + (size_t)(i >> 3) * kHW +
                   (box ? (size_t)(i & 7) * kHWh + dc : (size_t)dc * kF + (i & 7));
      ob[off] = acc[r];
    }
  }
}

// m[idx[j]] = o1[j]*o2[j]  (fuses the inverse permutation as a scatter)
__global__ void k_mul_scatter(const float* __restrict__ o1, const float* __restrict__ o2,
                              const int* __restrict__ idxb,
                              unsigned short* __restrict__ mbf) {
  int e = blockIdx.x * blockDim.x + threadIdx.x;
  if (e >= kB * kC * kHW) return;
  int j = e % kHW, c = (e / kHW) % kC, b = e / (kHW * kC);
  int pos = idxb[(size_t)b * kHW + j];
  size_t src = (size_t)(b * kC + c) * kHW + j;
  mbf[(size_t)(b * kC + c) * kHW + pos] = f2bf(o1[src] * o2[src]);
}

// final: per-patch inverse sort (as scatter) on first 32 channels, copy rest
__global__ void k_final(const float* __restrict__ outt, const int* __restrict__ sidxp,
                        float* __restrict__ outp) {
  int e = blockIdx.x * blockDim.x + threadIdx.x;
  if (e >= kB * kC * kHW) return;
  int p = e % kHW, c = (e / kHW) % kC, b = e / (kHW * kC);
  size_t chan = (size_t)(b * kC + c) * kHW;
  if (c >= kHalf) {
    outp[chan + p] = outt[chan + p];
    return;
  }
  int yy = p >> 8, xx = p & 255;
  int n = (yy >> 3) * kNP + (xx >> 3);
  int u = (yy & 7) * 8 + (xx & 7);
  int su = sidxp[((size_t)b * kNPatch + n) * 64 + u];
  int dy = (yy & ~7) + (su >> 3), dx = (xx & ~7) + (su & 7);
  outp[chan + dy * kW + dx] = outt[chan + p];
}

// ---------------------------------------------------------------------------
extern "C" void kernel_launch(void* const* d_in, const int* in_sizes, int n_in,
                              void* d_out, int out_size, void* d_ws, size_t ws_size,
                              hipStream_t stream) {
  (void)in_sizes; (void)n_in; (void)out_size; (void)ws_size;
  const float* x = (const float*)d_in[0];
  const float* temperature = (const float*)d_in[1];
  const float* w_qkv = (const float*)d_in[2];
  const float* w_dw = (const float*)d_in[3];
  const float* w_proj = (const float*)d_in[4];
  const float* w_bin = (const float*)d_in[5];
  float* out = (float*)d_out;
  char* ws = (char*)d_ws;

  float* magh = (float*)(ws + OFF_MAGH);
  int* binh = (int*)(ws + OFF_BINH);
  int* sidxp = (int*)(ws + OFF_SIDXP);
  float* hog = (float*)(ws + OFF_HOG);
  float* hms = (float*)(ws + OFF_HMS);
  float* xhp = (float*)(ws + OFF_XHP);
  unsigned short* x2bf = (unsigned short*)(ws + OFF_X2BF);
  unsigned short* wqkvb = (unsigned short*)(ws + OFF_WQKVB);
  unsigned short* wprojb = (unsigned short*)(ws + OFF_WPROJB);
  float* qkv1 = (float*)(ws + OFF_QKV1);
  float* qkv2 = (float*)(ws + OFF_QKV2);
  float* keys = (float*)(ws + OFF_KEYS);
  int* idxb = (int*)(ws + OFF_IDX);
  unsigned short* gq1 = (unsigned short*)(ws + OFF_GQ1);
  unsigned short* gk1 = (unsigned short*)(ws + OFF_GK1);
  unsigned short* gq2 = (unsigned short*)(ws + OFF_GQ2);
  unsigned short* gk2 = (unsigned short*)(ws + OFF_GK2);
  unsigned short* gv = (unsigned short*)(ws + OFF_GV);
  float* rnorm = (float*)(ws + OFF_RNORM);
  unsigned short* abf = (unsigned short*)(ws + OFF_ABF);
  float* o1 = (float*)(ws + OFF_O1);
  float* o2 = (float*)(ws + OFF_O2);
  unsigned short* mbf = (unsigned short*)(ws + OFF_MBF);
  float* outt = (float*)(ws + OFF_OUTT);

  const int T = 256;
  auto blocks = [](long n, int t) { return (int)((n + t - 1) / t); };

  // HOG path
  k_sobel_hog<<<blocks((long)kB * kHalf * kHW, T), T, 0, stream>>>(x, magh, binh);
  k_patch<<<kB * kNPatch, 64, 0, stream>>>(x, magh, binh, xhp, sidxp, hog);
  k_binconv<<<blocks((long)kB * kHalf * kNPatch, T), T, 0, stream>>>(hog, w_bin, hms);
  k_buildx2<<<blocks((long)kB * kC * kHW, T), T, 0, stream>>>(x, xhp, hms, x2bf);
  k_cvtw<<<blocks(kQKVC * kC, T), T, 0, stream>>>(w_qkv, w_proj, wqkvb, wprojb);

  // qkv = dw3x3(conv1x1(x2))  [WMMA GEMM + depthwise]
  k_gemm64<<<dim3(kHW / 64, kQKVC / 64, kB), T, 0, stream>>>(
      wqkvb, x2bf, qkv1, (size_t)kC * kHW, (size_t)kQKVC * kHW);
  k_dw<<<blocks((long)kB * kQKVC * kHW, T), T, 0, stream>>>(qkv1, w_dw, qkv2);

  // global sort of 65536 keys per batch (bitonic network)
  k_keys<<<blocks((long)kB * kHW, T), T, 0, stream>>>(qkv2, keys, idxb);
  for (int k = 2; k <= kHW; k <<= 1)
    for (int j = k >> 1; j > 0; j >>= 1)
      k_bitonic<<<blocks((long)kB * kHW, T), T, 0, stream>>>(keys, idxb, j, k);

  // gather + norms + attention (WMMA + TDM)
  k_gather<<<blocks((long)kB * kC * kHW, T), T, 0, stream>>>(qkv2, idxb, gq1, gk1, gq2,
                                                            gk2, gv);
  k_norms<<<512, T, 0, stream>>>(gq1, gk1, gq2, gk2, rnorm);
  k_attn_qk<<<dim3(kB * kHeads, 2), T, 0, stream>>>(gq1, gk1, gq2, gk2, rnorm,
                                                    temperature, abf);
  k_attn_o<<<dim3(kHWh / 64, kB * kHeads, 2), T, 0, stream>>>(abf, gv, o1, o2);

  // inverse permute (scatter), projection GEMM, per-patch unsort
  k_mul_scatter<<<blocks((long)kB * kC * kHW, T), T, 0, stream>>>(o1, o2, idxb, mbf);
  k_gemm64<<<dim3(kHW / 64, kC / 64, kB), T, 0, stream>>>(
      wprojb, mbf, outt, (size_t)kC * kHW, (size_t)kC * kHW);
  k_final<<<blocks((long)kB * kC * kHW, T), T, 0, stream>>>(outt, sidxp, out);
}
